// MedSAM_Encoder_34144990003609
// MI455X (gfx1250) — matrix operand order repaired and checked by reading, not compiled, not run on hardware
//
#include <hip/hip_runtime.h>
#include <math.h>

// ---------------------------------------------------------------------------
// MedSAM / SAM ViT-B encoder for gfx1250 (MI455X).
// All GEMM-like work runs through v_wmma_f32_16x16x32_bf16 (fp32 accumulate).
// Attention is flash-style (online softmax) so global blocks never
// materialize the 800MB score tensor.
// ---------------------------------------------------------------------------

typedef __attribute__((ext_vector_type(16))) __bf16 v16bf;
typedef __attribute__((ext_vector_type(4)))  __bf16 v4bf;
typedef __attribute__((ext_vector_type(8)))  float  v8f;

union FragBF {
  v16bf v;
  unsigned int u[8];
};
union Pack4 {
  v4bf h;
  uint2 u2;
};

__device__ __forceinline__ unsigned short f2bf(float x) {
  union { __bf16 h; unsigned short u; } c;
  c.h = (__bf16)x;
  return c.u;
}
// Pack a float4 into 4 bf16 (two v_cvt_pk_bf16_f32 + single b64 store path).
__device__ __forceinline__ uint2 pack_bf4(float4 f) {
  Pack4 p;
  p.h[0] = (__bf16)f.x;
  p.h[1] = (__bf16)f.y;
  p.h[2] = (__bf16)f.z;
  p.h[3] = (__bf16)f.w;
  return p.u2;
}
__device__ __forceinline__ unsigned int pack_bf2(float a, float b) {
  union { __bf16 h[2]; unsigned int u; } c;
  c.h[0] = (__bf16)a;
  c.h[1] = (__bf16)b;
  return c.u;
}
// A-matrix (16x32 bf16) K offsets chosen here (contiguous, not ISA table order): VGPR v holds K pair
// 2v + (lane>=16 ? 16 : 0)
__device__ __forceinline__ int patA(int v, int hi) {
  return 2 * v + hi * 16;
}
// B-matrix (32x16 bf16), same chosen order as A: VGPR v holds K pair 2v + (lane>=16 ? 16 : 0)

// ---------------------------------------------------------------------------
// Generic bf16-WMMA GEMM: C[M,N] = act(A[M,K] * W[N,K]^T + bias) + res
// A and W are fp32 in memory; converted to bf16 while staging into LDS
// (float4 loads, branchless clamp, packed bf16 stores).
// N multiple of 64, K multiple of 32. M may be ragged. act: 0 none, 1 GELU.
// ---------------------------------------------------------------------------
__global__ __launch_bounds__(128) void gemm_bf16_wmma(
    const float* __restrict__ A, const float* __restrict__ W,
    const float* __restrict__ bias, const float* __restrict__ res,
    float* __restrict__ C, int M, int N, int K, int act) {
  __shared__ unsigned short As[64][40];
  __shared__ unsigned short Bs[64][40];
  const int tid  = threadIdx.x;
  const int wave = tid >> 5, lane = tid & 31;
  const int hi   = lane >> 4, l16 = lane & 15;
  const int m0   = blockIdx.y * 64, n0 = blockIdx.x * 64;

  v8f acc[4];
#pragma unroll
  for (int s = 0; s < 4; ++s)
#pragma unroll
    for (int r = 0; r < 8; ++r) acc[s][r] = 0.f;

  for (int kc = 0; kc < K; kc += 32) {
    __syncthreads();
#pragma unroll
    for (int j = 0; j < 4; ++j) {
      int g = tid + 128 * j;           // 512 float4 groups per 64x32 tile
      int r = g >> 3, cg = (g & 7) * 4;
      // A tile (ragged M: clamp row, zero value -> v_cndmask, no branches)
      int gm = m0 + r;
      int gmc = gm < M ? gm : M - 1;
      float4 a4 = *(const float4*)&A[(size_t)gmc * K + kc + cg];
      if (gm >= M) { a4.x = 0.f; a4.y = 0.f; a4.z = 0.f; a4.w = 0.f; }
      *(uint2*)&As[r][cg] = pack_bf4(a4);
      // W tile (N always a multiple of 64)
      float4 b4 = *(const float4*)&W[(size_t)(n0 + r) * K + kc + cg];
      *(uint2*)&Bs[r][cg] = pack_bf4(b4);
      if (j == 0 && kc + 32 < K)  // global_prefetch_b8 of next K slice
        __builtin_prefetch(&W[(size_t)(n0 + r) * K + kc + 32 + cg], 0, 1);
    }
    __syncthreads();
    // Preload A frag + all four B frags, then issue the WMMAs back-to-back
    // (one dscnt wait instead of four).
    FragBF a;
    {
      int row = wave * 16 + l16;
#pragma unroll
      for (int v2 = 0; v2 < 8; ++v2)
        a.u[v2] = *(const unsigned int*)&As[row][patA(v2, hi)];
    }
    FragBF b[4];
#pragma unroll
    for (int s = 0; s < 4; ++s) {
      int cn = s * 16 + l16;
#pragma unroll
      for (int v2 = 0; v2 < 8; ++v2)
        b[s].u[v2] = *(const unsigned int*)&Bs[cn][2 * v2 + hi * 16];
    }
#pragma unroll
    for (int s = 0; s < 4; ++s)
      acc[s] = __builtin_amdgcn_wmma_f32_16x16x32_bf16(
          false, a.v, false, b[s].v, (short)0, acc[s], false, false);
  }
#pragma unroll
  for (int s = 0; s < 4; ++s) {
#pragma unroll
    for (int r = 0; r < 8; ++r) {
      int gm = m0 + wave * 16 + r + hi * 8;
      if (gm < M) {
        int gn = n0 + s * 16 + l16;
        float v = acc[s][r];
        if (bias) v += bias[gn];
        if (act == 1) v = 0.5f * v * (1.f + erff(v * 0.70710678118f));
        if (res) v += res[(size_t)gm * N + gn];
        C[(size_t)gm * N + gn] = v;
      }
    }
  }
}

// ---------------------------------------------------------------------------
// Row LayerNorm (also implements LayerNorm2d on [HW, C] layout).
// One wave per row; C <= 768.
// ---------------------------------------------------------------------------
__global__ __launch_bounds__(256) void ln_rows(
    const float* __restrict__ x, const float* __restrict__ w,
    const float* __restrict__ b, float* __restrict__ out, int rows, int C) {
  int wave = threadIdx.x >> 5, lane = threadIdx.x & 31;
  int row = blockIdx.x * 8 + wave;
  if (row >= rows) return;
  const float* xr = x + (size_t)row * C;
  float s = 0.f, s2 = 0.f;
  for (int c = lane; c < C; c += 32) { float v = xr[c]; s += v; s2 += v * v; }
  for (int off = 16; off > 0; off >>= 1) {
    s  += __shfl_xor(s,  off, 32);
    s2 += __shfl_xor(s2, off, 32);
  }
  float mu = s / C;
  float var = s2 / C - mu * mu;
  float rstd = rsqrtf(var + 1e-6f);
  float* orow = out + (size_t)row * C;
  for (int c = lane; c < C; c += 32) orow[c] = (xr[c] - mu) * rstd * w[c] + b[c];
}

// ---------------------------------------------------------------------------
// Flash attention (one wave = one 16-query tile of one (window,head)).
// q/k/v: [BH, N, 64] fp32. rel_h/rel_w: [BH, N, side] fp32 bias tables.
// out: [nW*N, 768] with head h written at column h*64.
// ---------------------------------------------------------------------------
__global__ __launch_bounds__(128) void attn_flash(
    const float* __restrict__ Q, const float* __restrict__ K,
    const float* __restrict__ V, const float* __restrict__ RH,
    const float* __restrict__ RW, float* __restrict__ out, int N, int side) {
  __shared__ unsigned short Kt[32][72];      // [key][d]
  __shared__ unsigned short Vt[64][40];      // [d][key] (transposed)
  __shared__ unsigned short Pt[4][16][40];   // per-wave P tile [row][key32]
  const int tid = threadIdx.x, wave = tid >> 5, lane = tid & 31;
  const int hi = lane >> 4, l16 = lane & 15;
  const int bh = blockIdx.y;
  const int win = bh / 12, head = bh % 12;
  const int q0 = blockIdx.x * 64 + wave * 16;
  const float scale = 0.125f;  // 64^-0.5
  const float* qb = Q + (size_t)bh * N * 64;
  const float* kb = K + (size_t)bh * N * 64;
  const float* vb = V + (size_t)bh * N * 64;

  FragBF qf[2];
  {
    int qi = q0 + l16;
    bool okq = qi < N;
    int qic = okq ? qi : N - 1;
#pragma unroll
    for (int c = 0; c < 2; ++c)
#pragma unroll
      for (int v2 = 0; v2 < 8; ++v2) {
        int d0 = patA(v2, hi) + c * 32;
        unsigned int pk = pack_bf2(qb[(size_t)qic * 64 + d0],
                                   qb[(size_t)qic * 64 + d0 + 1]);
        qf[c].u[v2] = okq ? pk : 0u;
      }
  }

  v8f o[4];
#pragma unroll
  for (int f = 0; f < 4; ++f)
#pragma unroll
    for (int r = 0; r < 8; ++r) o[f][r] = 0.f;
  float rowM[8], rowL[8];
#pragma unroll
  for (int r = 0; r < 8; ++r) { rowM[r] = -1e30f; rowL[r] = 0.f; }

  for (int kc = 0; kc < N; kc += 32) {
    __syncthreads();
#pragma unroll
    for (int j = 0; j < 4; ++j) {
      int g = tid + 128 * j;             // 512 float4 groups per 32x64 tile
      int key = g >> 4, dg = (g & 15) * 4;
      int gk = kc + key;
      int gkc = gk < N ? gk : N - 1;
      float4 k4 = *(const float4*)&kb[(size_t)gkc * 64 + dg];
      float4 v4 = *(const float4*)&vb[(size_t)gkc * 64 + dg];
      if (gk >= N) {
        k4.x = 0.f; k4.y = 0.f; k4.z = 0.f; k4.w = 0.f;
        v4.x = 0.f; v4.y = 0.f; v4.z = 0.f; v4.w = 0.f;
      }
      *(uint2*)&Kt[key][dg] = pack_bf4(k4);
      Vt[dg + 0][key] = f2bf(v4.x);
      Vt[dg + 1][key] = f2bf(v4.y);
      Vt[dg + 2][key] = f2bf(v4.z);
      Vt[dg + 3][key] = f2bf(v4.w);
    }
    __syncthreads();

    float S2[2][8];
#pragma unroll
    for (int sub = 0; sub < 2; ++sub) {
      // preload both K frags (d-chunks), then two chained WMMAs
      FragBF kf[2];
      int cn = sub * 16 + l16;
#pragma unroll
      for (int c = 0; c < 2; ++c)
#pragma unroll
        for (int v2 = 0; v2 < 8; ++v2)
          kf[c].u[v2] = *(const unsigned int*)&Kt[cn][2 * v2 + hi * 16 + c * 32];
      v8f s;
#pragma unroll
      for (int r = 0; r < 8; ++r) s[r] = 0.f;
      s = __builtin_amdgcn_wmma_f32_16x16x32_bf16(
          false, qf[0].v, false, kf[0].v, (short)0, s, false, false);
      s = __builtin_amdgcn_wmma_f32_16x16x32_bf16(
          false, qf[1].v, false, kf[1].v, (short)0, s, false, false);
#pragma unroll
      for (int r = 0; r < 8; ++r) {
        int key = kc + sub * 16 + l16;
        int qi = q0 + r + hi * 8;
        float val = s[r] * scale;
        if (key < N) {
          if (qi < N) {
            int kh = key / side, kw = key % side;
            size_t base = ((size_t)bh * N + qi) * side;
            val += RH[base + kh] + RW[base + kw];
          }
        } else {
          val = -1e30f;  // masked key
        }
        S2[sub][r] = val;
      }
    }
    // online softmax (rows live across 16 lanes of a half-wave)
#pragma unroll
    for (int r = 0; r < 8; ++r) {
      float m2 = fmaxf(S2[0][r], S2[1][r]);
      for (int off = 1; off < 16; off <<= 1) m2 = fmaxf(m2, __shfl_xor(m2, off, 32));
      float newM = fmaxf(rowM[r], m2);
      float alpha = __expf(rowM[r] - newM);
      rowM[r] = newM;
      float p0 = __expf(S2[0][r] - newM);
      float p1 = __expf(S2[1][r] - newM);
      float rs = p0 + p1;
      for (int off = 1; off < 16; off <<= 1) rs += __shfl_xor(rs, off, 32);
      rowL[r] = rowL[r] * alpha + rs;
#pragma unroll
      for (int f = 0; f < 4; ++f) o[f][r] *= alpha;
      int rl = r + hi * 8;                 // C-layout row -> LDS
      Pt[wave][rl][l16]      = f2bf(p0);
      Pt[wave][rl][16 + l16] = f2bf(p1);
    }
    // same-wave LDS RAW: wait for the P-tile stores, then reload in A-layout
    asm volatile("s_wait_dscnt 0" ::: "memory");
    FragBF pf;
#pragma unroll
    for (int v2 = 0; v2 < 8; ++v2)
      pf.u[v2] = *(const unsigned int*)&Pt[wave][l16][patA(v2, hi)];
    FragBF vf[4];
#pragma unroll
    for (int f = 0; f < 4; ++f) {
      int dc = f * 16 + l16;
#pragma unroll
      for (int v2 = 0; v2 < 8; ++v2)
        vf[f].u[v2] = *(const unsigned int*)&Vt[dc][2 * v2 + hi * 16];
    }
#pragma unroll
    for (int f = 0; f < 4; ++f)
      o[f] = __builtin_amdgcn_wmma_f32_16x16x32_bf16(
          false, pf.v, false, vf[f].v, (short)0, o[f], false, false);
  }
#pragma unroll
  for (int f = 0; f < 4; ++f)
#pragma unroll
    for (int r = 0; r < 8; ++r) {
      int qi = q0 + r + hi * 8;
      if (qi < N) {
        float val = o[f][r] / rowL[r];
        size_t tok = (size_t)win * N + qi;
        out[tok * 768 + head * 64 + f * 16 + l16] = val;
      }
    }
}

// ---------------------------------------------------------------------------
// Elementwise / reshape helpers
// ---------------------------------------------------------------------------
__global__ void im2col_patch_k(const float* __restrict__ x, float* __restrict__ im) {
  int total = 4096 * 768;
  for (int idx = blockIdx.x * 256 + threadIdx.x; idx < total; idx += gridDim.x * 256) {
    int p = idx / 768, j = idx % 768;
    int c = j >> 8, rr = j & 255, py = rr >> 4, px = rr & 15;
    int gy = p >> 6, gx = p & 63;
    im[idx] = x[(size_t)c * 1048576 + (size_t)(gy * 16 + py) * 1024 + (gx * 16 + px)];
  }
}

__global__ void gather_win_k(const float* __restrict__ ln, float* __restrict__ xw) {
  int total = 4900 * 768;
  for (int idx = blockIdx.x * 256 + threadIdx.x; idx < total; idx += gridDim.x * 256) {
    int t = idx / 768, c = idx % 768;
    int w = t / 196, n = t % 196;
    int wy = w / 5, wx = w % 5;
    int hy = wy * 14 + n / 14, hx = wx * 14 + n % 14;
    xw[idx] = (hy < 64 && hx < 64) ? ln[(size_t)(hy * 64 + hx) * 768 + c] : 0.f;
  }
}

__global__ void scatter_add_win_k(float* __restrict__ x, const float* __restrict__ y) {
  int total = 4096 * 768;
  for (int idx = blockIdx.x * 256 + threadIdx.x; idx < total; idx += gridDim.x * 256) {
    int t = idx / 768, c = idx % 768;
    int hy = t / 64, hx = t % 64;
    int w = (hy / 14) * 5 + (hx / 14);
    int n = (hy % 14) * 14 + (hx % 14);
    x[idx] += y[(size_t)(w * 196 + n) * 768 + c];
  }
}

__global__ void add_inplace_k(float* __restrict__ x, const float* __restrict__ y, int total) {
  for (int idx = blockIdx.x * 256 + threadIdx.x; idx < total; idx += gridDim.x * 256)
    x[idx] += y[idx];
}

__global__ void qkv_split_k(const float* __restrict__ qkv, float* __restrict__ Qo,
                            float* __restrict__ Ko, float* __restrict__ Vo,
                            int T, int N) {
  int total = T * 2304;
  for (int idx = blockIdx.x * 256 + threadIdx.x; idx < total; idx += gridDim.x * 256) {
    int t = idx / 2304, c = idx % 2304;
    int m = c / 768, rem = c % 768, h = rem / 64, d = rem % 64;
    int w = t / N, n = t % N;
    size_t dst = ((size_t)(w * 12 + h) * N + n) * 64 + d;
    float v = qkv[idx];
    if (m == 0) Qo[dst] = v; else if (m == 1) Ko[dst] = v; else Vo[dst] = v;
  }
}

__global__ void relpos_k(const float* __restrict__ q, const float* __restrict__ th,
                         const float* __restrict__ tw, float* __restrict__ rh,
                         float* __restrict__ rw, int BH, int N, int side) {
  int total = BH * N * side;
  for (int idx = blockIdx.x * 256 + threadIdx.x; idx < total; idx += gridDim.x * 256) {
    int k = idx % side; int t = idx / side; int n = t % N; int bh = t / N;
    int qh = n / side, qw = n % side;
    const float* qp = q + ((size_t)bh * N + n) * 64;
    const float* ph = th + (size_t)(qh - k + side - 1) * 64;
    const float* pw = tw + (size_t)(qw - k + side - 1) * 64;
    float ah = 0.f, aw = 0.f;
#pragma unroll 8
    for (int d = 0; d < 64; ++d) { float qv = qp[d]; ah += qv * ph[d]; aw += qv * pw[d]; }
    rh[idx] = ah; rw[idx] = aw;
  }
}

__global__ void im2col_neck_k(const float* __restrict__ n1, float* __restrict__ im) {
  int total = 4096 * 2304;
  for (int idx = blockIdx.x * 256 + threadIdx.x; idx < total; idx += gridDim.x * 256) {
    int p = idx / 2304, j = idx % 2304;
    int ci = j / 9, kk = j % 9;
    int py = p / 64 + kk / 3 - 1, px = p % 64 + kk % 3 - 1;
    im[idx] = (py >= 0 && py < 64 && px >= 0 && px < 64)
                  ? n1[(size_t)(py * 64 + px) * 256 + ci] : 0.f;
  }
}

__global__ __launch_bounds__(256) void meanpool_k(const float* __restrict__ x,
                                                  float* __restrict__ out) {
  __shared__ float red[256];
  int c = blockIdx.x;
  float s = 0.f;
  for (int r = threadIdx.x; r < 4096; r += 256) s += x[(size_t)r * 256 + c];
  red[threadIdx.x] = s; __syncthreads();
  for (int st = 128; st > 0; st >>= 1) {
    if (threadIdx.x < st) red[threadIdx.x] += red[threadIdx.x + st];
    __syncthreads();
  }
  if (threadIdx.x == 0) out[c] = red[0] / 4096.f;
}

// ---------------------------------------------------------------------------
// Host orchestration
// ---------------------------------------------------------------------------
static inline void launch_gemm(const float* A, const float* W, const float* bias,
                               const float* res, float* C, int M, int N, int K,
                               int act, hipStream_t s) {
  dim3 g(N / 64, (M + 63) / 64);
  gemm_bf16_wmma<<<g, 128, 0, s>>>(A, W, bias, res, C, M, N, K, act);
}
#define EWG(total) dim3(((total) + 255) / 256), dim3(256), 0, stream

extern "C" void kernel_launch(void* const* d_in, const int* in_sizes, int n_in,
                              void* d_out, int out_size, void* d_ws, size_t ws_size,
                              hipStream_t stream) {
  (void)in_sizes; (void)n_in; (void)out_size;
  // Input order: x, then params in dict insertion order, blocks last.
  auto P = [&](int i) -> const float* { return (const float*)d_in[i]; };
  const float* x_img     = P(0);
  const float* patch_w   = P(1);
  const float* patch_b   = P(2);
  const float* pos_embed = P(3);
  const float* neck_c1   = P(4);
  const float* neck_l1w  = P(5);
  const float* neck_l1b  = P(6);
  const float* neck_c2   = P(7);
  const float* neck_l2w  = P(8);
  const float* neck_l2b  = P(9);
  auto BP = [&](int blk, int j) -> const float* { return (const float*)d_in[10 + blk * 14 + j]; };

  // Workspace layout (floats), phase-overlapped. Total ~186 MB.
  float* ws = (float*)d_ws;
  const size_t oX = 0, oLN = 3145728, oXW = 6291456, oQKV = 10054656,
               oQ = 21344256, oK = 25107456, oV = 28870656, oRH = 32633856,
               oRW = 35779584, oAO = 38925312, oY = 42688512, oEND = 46451712;
  if (ws_size < oEND * sizeof(float)) return;  // workspace too small: bail out
  float* X   = ws + oX;    // [4096,768] running activations
  float* LN  = ws + oLN;   // ln1 / ln2 output
  float* XW  = ws + oXW;   // windowed tokens [4900,768]
  float* QKV = ws + oQKV;  // [T,2304]
  float* Qb  = ws + oQ;    // [BH,N,64]
  float* Kb  = ws + oK;
  float* Vb  = ws + oV;
  float* RH  = ws + oRH;   // rel bias tables  (neck: N1)
  float* RW  = ws + oRW;   //                  (neck: N2)
  float* AO  = ws + oAO;   // attention out merged [T,768]
  float* Y   = ws + oY;    // proj out [T,768]
  float* H   = ws + oQKV;  // MLP hidden [4096,3072] (QKV region, phase-disjoint)
  float* IM  = ws + oQKV;  // im2col scratch (phase-disjoint)
  float* N1  = ws + oRH;
  float* N2  = ws + oRW;

  // ---- patch embed: im2col + GEMM (+bias, +pos_embed as residual) ----
  im2col_patch_k<<<EWG(4096 * 768)>>>(x_img, IM);
  launch_gemm(IM, patch_w, patch_b, pos_embed, X, 4096, 768, 768, 0, stream);

  // ---- 12 transformer blocks ----
  for (int i = 0; i < 12; ++i) {
    bool global = (i == 2 || i == 5 || i == 8 || i == 11);
    int NW = global ? 1 : 25;      // windows
    int Nn = global ? 4096 : 196;  // tokens per window
    int side = global ? 64 : 14;
    int T = NW * Nn;
    int BH = NW * 12;

    ln_rows<<<512, 256, 0, stream>>>(X, BP(i, 0), BP(i, 1), LN, 4096, 768);
    const float* Aattn = LN;
    if (!global) {
      gather_win_k<<<EWG(4900 * 768)>>>(LN, XW);
      Aattn = XW;
    }
    launch_gemm(Aattn, BP(i, 2), BP(i, 3), nullptr, QKV, T, 2304, 768, 0, stream);
    qkv_split_k<<<EWG(T * 2304)>>>(QKV, Qb, Kb, Vb, T, Nn);
    relpos_k<<<EWG(BH * Nn * side)>>>(Qb, BP(i, 6), BP(i, 7), RH, RW, BH, Nn, side);
    attn_flash<<<dim3((Nn + 63) / 64, BH), 128, 0, stream>>>(Qb, Kb, Vb, RH, RW, AO, Nn, side);
    launch_gemm(AO, BP(i, 4), BP(i, 5), nullptr, Y, T, 768, 768, 0, stream);
    if (!global)
      scatter_add_win_k<<<EWG(4096 * 768)>>>(X, Y);
    else
      add_inplace_k<<<EWG(4096 * 768)>>>(X, Y, 4096 * 768);

    // MLP: LN2 -> fc1(+GELU) -> fc2(+residual into X)
    ln_rows<<<512, 256, 0, stream>>>(X, BP(i, 8), BP(i, 9), LN, 4096, 768);
    launch_gemm(LN, BP(i, 10), BP(i, 11), nullptr, H, 4096, 3072, 768, 1, stream);
    launch_gemm(H, BP(i, 12), BP(i, 13), X, X, 4096, 768, 3072, 0, stream);
  }

  // ---- neck: 1x1 conv -> LN2d -> 3x3 conv -> LN2d -> global mean ----
  launch_gemm(X, neck_c1, nullptr, nullptr, N1, 4096, 256, 768, 0, stream);
  ln_rows<<<512, 256, 0, stream>>>(N1, neck_l1w, neck_l1b, N1, 4096, 256);
  im2col_neck_k<<<EWG(4096 * 2304)>>>(N1, IM);
  launch_gemm(IM, neck_c2, nullptr, nullptr, N2, 4096, 256, 2304, 0, stream);
  ln_rows<<<512, 256, 0, stream>>>(N2, neck_l2w, neck_l2b, N2, 4096, 256);
  meanpool_k<<<256, 256, 0, stream>>>(N2, (float*)d_out);
}
